// AdaptiveResBlock_53094385713974
// MI455X (gfx1250) — matrix operand order repaired and checked by reading, not compiled, not run on hardware
//
#include <hip/hip_runtime.h>
#include <stdint.h>

// ---------------- problem constants (match reference) ----------------
#define NB    8        // batch
#define CCH   256      // channels
#define TLEN  8192     // time
#define LSLOPE 0.1f

// ---------------- tiling ----------------
#define TT    64                   // time steps per block
#define OCT   64                   // output channels per block
#define KC    64                   // input-channel chunk staged in LDS
#define NCHUNK (CCH / KC)          // 4
#define HALO  64                   // max |round(d*dil)| (d<16, dil<=4)
#define HLEN  (TT + 2*HALO)        // 192 haloed times (stage 1)
#define YLEN  (TT + 2)             // +/-1 halo (stage 2, conv3)
#define KSTR  72                   // activation LDS row stride [t][c], 144B
#define WSS   72                   // weight LDS row stride [o][c], 144B

typedef __attribute__((ext_vector_type(16))) __bf16         v16bf;
typedef __attribute__((ext_vector_type(16))) unsigned short v16u;
typedef __attribute__((ext_vector_type(8)))  unsigned short v8u;
typedef __attribute__((ext_vector_type(8)))  float          v8f;

__device__ __forceinline__ float lrelu(float v) { return v > 0.f ? v : LSLOPE * v; }

__device__ __forceinline__ unsigned short f2bf(float f) {
    return __builtin_bit_cast(unsigned short, (__bf16)f);
}
__device__ __forceinline__ v8u lds8(const unsigned short* p) {
    return *reinterpret_cast<const v8u*>(p);
}
__device__ __forceinline__ v16u cat8(v8u lo, v8u hi) {
    return __builtin_shufflevector(lo, hi, 0,1,2,3,4,5,6,7,8,9,10,11,12,13,14,15);
}
__device__ __forceinline__ v8f wmma_bf16(v16u a, v16u b, v8f c) {
    return __builtin_amdgcn_wmma_f32_16x16x32_bf16(
        false, __builtin_bit_cast(v16bf, a),
        false, __builtin_bit_cast(v16bf, b),
        (short)0, c, false, false);
}

// ---- CDNA5 async global->LDS copy (ASYNCcnt path) ----
__device__ __forceinline__ uint32_t lds_off(const void* p) {
    // generic pointers to LDS carry the LDS byte address in the low 32 bits
    return (uint32_t)(uintptr_t)p;
}
__device__ __forceinline__ void async_cp16(uint32_t lo, const void* g) {
    asm volatile("global_load_async_to_lds_b128 %0, %1, off"
                 :: "v"(lo), "v"((uint64_t)(uintptr_t)g) : "memory");
}
__device__ __forceinline__ void wait_async0() {
    asm volatile("s_wait_asynccnt 0x0" ::: "memory");
}

// =====================================================================
// Prep A: bf16 weight copies.  wcb/wpb/wfb: [dil][o][c].  wab: [dil][tap][o][c].
// =====================================================================
__global__ __launch_bounds__(256)
void prep_weights(const float* __restrict__ WC, const float* __restrict__ WP,
                  const float* __restrict__ WF, const float* __restrict__ WA,
                  unsigned short* __restrict__ wcb, unsigned short* __restrict__ wpb,
                  unsigned short* __restrict__ wfb, unsigned short* __restrict__ wab)
{
    const int idx = blockIdx.x * 256 + threadIdx.x;      // [0, 3*65536)
    const int dil = idx >> 16, rem = idx & 0xFFFF;       // rem = o*256 + c
    wcb[idx] = f2bf(WC[idx]);
    wpb[idx] = f2bf(WP[idx]);
    wfb[idx] = f2bf(WF[idx]);
#pragma unroll
    for (int tap = 0; tap < 3; ++tap)
        wab[((size_t)(dil * 3 + tap) << 16) + rem] =
            f2bf(WA[(size_t)dil * 3 * 65536 + (size_t)rem * 3 + tap]);
}

// =====================================================================
// Prep B: xbf[b][t][c] = bf16(lrelu(x[b][c][t]))  (LDS-tiled transpose)
// =====================================================================
__global__ __launch_bounds__(256)
void prep_x(const float* __restrict__ x, unsigned short* __restrict__ xbf)
{
    __shared__ alignas(16) unsigned short tile[64 * KSTR];   // [t][c]
    const int tb = blockIdx.x * 64, cb = blockIdx.y * 64, b = blockIdx.z;
    const int tid = threadIdx.x;
    for (int e = tid; e < 64 * 16; e += 256) {               // 64 ch rows x 16 float4
        const int ch = e >> 4, t4 = (e & 15) * 4;
        const float4 xv = *reinterpret_cast<const float4*>(
            &x[((size_t)b * CCH + cb + ch) * TLEN + tb + t4]);
        tile[(t4 + 0) * KSTR + ch] = f2bf(lrelu(xv.x));
        tile[(t4 + 1) * KSTR + ch] = f2bf(lrelu(xv.y));
        tile[(t4 + 2) * KSTR + ch] = f2bf(lrelu(xv.z));
        tile[(t4 + 3) * KSTR + ch] = f2bf(lrelu(xv.w));
    }
    __syncthreads();
    for (int v = tid; v < 64 * 8; v += 256) {                // 64 t rows x 8 v8u
        const int tl = v >> 3, c8 = (v & 7) * 8;
        *reinterpret_cast<v8u*>(&xbf[((size_t)b * TLEN + tb + tl) * CCH + cb + c8]) =
            lds8(&tile[tl * KSTR + c8]);
    }
}

// =====================================================================
// Stage 1: y = lrelu( WC@xt + WP@gatherP + WF@gatherF + biases )
// xt = xbf (bf16 lrelu'd activations, [b][t][c]);  y: [b][t][c] bf16.
// Double-buffered LDS, async global->LDS staging overlapped with WMMA.
// =====================================================================
__global__ __launch_bounds__(256)
void arb_stage1(const unsigned short* __restrict__ xbf, const float* __restrict__ d,
                const unsigned short* __restrict__ wc, const unsigned short* __restrict__ wp,
                const unsigned short* __restrict__ wf,
                const float* __restrict__ bC, const float* __restrict__ bP,
                const float* __restrict__ bF,
                unsigned short* __restrict__ y, int dilation)
{
    __shared__ alignas(16) unsigned short xs[2][(HLEN + 1) * KSTR]; // + zero row
    __shared__ alignas(16) unsigned short swc[2][OCT * WSS];
    __shared__ alignas(16) unsigned short swp[2][OCT * WSS];
    __shared__ alignas(16) unsigned short swf[2][OCT * WSS];
    __shared__ int sidxP[TT];
    __shared__ int sidxF[TT];

    const int tb = blockIdx.x, b = blockIdx.y, ob = blockIdx.z;
    const int t0 = tb * TT, obase = ob * OCT;
    const int tid = threadIdx.x;
    const bool interiorT = (t0 >= HALO) && (t0 + TT + HALO <= TLEN);

    // per-time gather rows; invalid -> zero row (HLEN)
    if (tid < TT) {
        const int tg = t0 + tid;
        const float off = d[(size_t)b * TLEN + tg] * (float)dilation;
        const int ip = (int)rintf((float)tg - off);
        const int iF = (int)rintf((float)tg + off);
        int lp = ip - (t0 - HALO);
        int lf = iF - (t0 - HALO);
        lp = lp < 0 ? 0 : (lp > HLEN - 1 ? HLEN - 1 : lp);
        lf = lf < 0 ? 0 : (lf > HLEN - 1 ? HLEN - 1 : lf);
        sidxP[tid] = (ip >= 0)   ? lp : HLEN;
        sidxF[tid] = (iF < TLEN) ? lf : HLEN;
    }
    if (tid < 16) {                              // zero rows (both buffers)
        v8u z;
#pragma unroll
        for (int i = 0; i < 8; ++i) z[i] = 0;
        *reinterpret_cast<v8u*>(&xs[tid >> 3][HLEN * KSTR + (tid & 7) * 8]) = z;
    }

    // issue (or sync-copy on edge blocks) one chunk's staging into buffer bi
    auto issueChunk = [&](int kc, int bi) {
        unsigned short* xsb = xs[bi];
        if (interiorT) {
            for (int v = tid; v < 8 * HLEN; v += 256) {
                const int tl = v >> 3, c8 = (v & 7) * 8;
                const int gt = t0 - HALO + tl;
                async_cp16(lds_off(&xsb[tl * KSTR + c8]),
                           &xbf[((size_t)b * TLEN + gt) * CCH + kc + c8]);
            }
        } else {
            for (int v = tid; v < 8 * HLEN; v += 256) {
                const int tl = v >> 3, c8 = (v & 7) * 8;
                const int gt = t0 - HALO + tl;
                v8u val;
                if (gt >= 0 && gt < TLEN)
                    val = *reinterpret_cast<const v8u*>(
                        &xbf[((size_t)b * TLEN + gt) * CCH + kc + c8]);
                else {
#pragma unroll
                    for (int i = 0; i < 8; ++i) val[i] = 0;
                }
                *reinterpret_cast<v8u*>(&xsb[tl * KSTR + c8]) = val;
            }
        }
        for (int v = tid; v < 8 * OCT; v += 256) {
            const int o = v >> 3, c8 = (v & 7) * 8;
            const size_t gi = (size_t)(obase + o) * CCH + kc + c8;
            async_cp16(lds_off(&swc[bi][o * WSS + c8]), &wc[gi]);
            async_cp16(lds_off(&swp[bi][o * WSS + c8]), &wp[gi]);
            async_cp16(lds_off(&swf[bi][o * WSS + c8]), &wf[gi]);
        }
    };

    const int wv = tid >> 5, ln = tid & 31;
    const int l16 = ln & 15, half = ln >> 4;
    const int mt = wv & 3, nt0 = wv >> 2;

    const int wrow   = (mt * 16 + l16) * WSS;
    const int aoffLo = 16 * half;
    const int aoffHi = 16 * half + 8;
    const int boff   = 16 * half;

    int rowC[2], rowP[2], rowF[2];
#pragma unroll
    for (int p = 0; p < 2; ++p) {
        const int tl = (nt0 + 2 * p) * 16 + l16;
        rowC[p] = (HALO + tl) * KSTR;
        // sidx written by this block's threads before first barrier below
        rowP[p] = 0; rowF[p] = 0;
    }

    v8f acc0, acc1;
#pragma unroll
    for (int i = 0; i < 8; ++i) { acc0[i] = 0.f; acc1[i] = 0.f; }

    issueChunk(0, 0);

#pragma unroll
    for (int ci = 0; ci < NCHUNK; ++ci) {
        const int bi = ci & 1;
        wait_async0();
        __syncthreads();
        if (ci == 0) {       // sidx now visible
#pragma unroll
            for (int p = 0; p < 2; ++p) {
                const int tl = (nt0 + 2 * p) * 16 + l16;
                rowP[p] = sidxP[tl] * KSTR;
                rowF[p] = sidxF[tl] * KSTR;
            }
        }
        if (ci + 1 < NCHUNK) issueChunk((ci + 1) * KC, bi ^ 1);

        const unsigned short* xsb = xs[bi];
        const unsigned short* wcb = swc[bi];
        const unsigned short* wpb = swp[bi];
        const unsigned short* wfb = swf[bi];
#pragma unroll
        for (int kk = 0; kk < 2; ++kk) {
            const int kkb = kk * 32;
            const v16u ac = cat8(lds8(wcb + wrow + kkb + aoffLo), lds8(wcb + wrow + kkb + aoffHi));
            const v16u ap = cat8(lds8(wpb + wrow + kkb + aoffLo), lds8(wpb + wrow + kkb + aoffHi));
            const v16u af = cat8(lds8(wfb + wrow + kkb + aoffLo), lds8(wfb + wrow + kkb + aoffHi));
            const int bo = kkb + boff;
            const v16u bc0 = cat8(lds8(xsb + rowC[0] + bo), lds8(xsb + rowC[0] + bo + 8));
            const v16u bc1 = cat8(lds8(xsb + rowC[1] + bo), lds8(xsb + rowC[1] + bo + 8));
            const v16u bp0 = cat8(lds8(xsb + rowP[0] + bo), lds8(xsb + rowP[0] + bo + 8));
            const v16u bp1 = cat8(lds8(xsb + rowP[1] + bo), lds8(xsb + rowP[1] + bo + 8));
            const v16u bf0 = cat8(lds8(xsb + rowF[0] + bo), lds8(xsb + rowF[0] + bo + 8));
            const v16u bf1 = cat8(lds8(xsb + rowF[1] + bo), lds8(xsb + rowF[1] + bo + 8));
            acc0 = wmma_bf16(ac, bc0, acc0);
            acc1 = wmma_bf16(ac, bc1, acc1);
            acc0 = wmma_bf16(ap, bp0, acc0);
            acc1 = wmma_bf16(ap, bp1, acc1);
            acc0 = wmma_bf16(af, bf0, acc0);
            acc1 = wmma_bf16(af, bf1, acc1);
        }
    }

    // epilogue: +(bC+bP+bF), lrelu, one b128 store per tile
    const int ob8 = obase + mt * 16 + 8 * half;
    const float4 c0 = *reinterpret_cast<const float4*>(&bC[ob8]);
    const float4 c1 = *reinterpret_cast<const float4*>(&bC[ob8 + 4]);
    const float4 p0 = *reinterpret_cast<const float4*>(&bP[ob8]);
    const float4 p1 = *reinterpret_cast<const float4*>(&bP[ob8 + 4]);
    const float4 f0 = *reinterpret_cast<const float4*>(&bF[ob8]);
    const float4 f1 = *reinterpret_cast<const float4*>(&bF[ob8 + 4]);
    const float bias[8] = { c0.x + p0.x + f0.x, c0.y + p0.y + f0.y,
                            c0.z + p0.z + f0.z, c0.w + p0.w + f0.w,
                            c1.x + p1.x + f1.x, c1.y + p1.y + f1.y,
                            c1.z + p1.z + f1.z, c1.w + p1.w + f1.w };
#pragma unroll
    for (int p = 0; p < 2; ++p) {
        const v8f acc = p ? acc1 : acc0;
        const int t = t0 + (nt0 + 2 * p) * 16 + l16;
        v8u pk;
#pragma unroll
        for (int r = 0; r < 8; ++r) pk[r] = f2bf(lrelu(acc[r] + bias[r]));
        *reinterpret_cast<v8u*>(&y[((size_t)b * TLEN + t) * CCH + ob8]) = pk;
    }
}

// =====================================================================
// Stage 2: v = conv3(y, WA) + bA + x_in; x_out = v (fp32 [b][c][t]);
// xbf_out = bf16(lrelu(v)) ([b][t][c], next layer's stage-1 input).
// =====================================================================
__global__ __launch_bounds__(256)
void arb_stage2(const unsigned short* __restrict__ y,
                const unsigned short* __restrict__ wa,   // [tap][o][c] bf16
                const float* __restrict__ bA,
                const float* __restrict__ xin, float* __restrict__ xout,
                unsigned short* __restrict__ xbf_out, int write_xbf)
{
    __shared__ alignas(16) unsigned short ys[2][YLEN * KSTR];
    __shared__ alignas(16) unsigned short swa[2][3][OCT * WSS];

    const int tb = blockIdx.x, b = blockIdx.y, ob = blockIdx.z;
    const int t0 = tb * TT, obase = ob * OCT;
    const int tid = threadIdx.x;
    const bool interiorT = (t0 >= 1) && (t0 + TT + 1 <= TLEN);

    auto issueChunk = [&](int kc, int bi) {
        unsigned short* ysb = ys[bi];
        if (interiorT) {
            for (int v = tid; v < 8 * YLEN; v += 256) {
                const int tl = v >> 3, c8 = (v & 7) * 8;
                const int gt = t0 - 1 + tl;
                async_cp16(lds_off(&ysb[tl * KSTR + c8]),
                           &y[((size_t)b * TLEN + gt) * CCH + kc + c8]);
            }
        } else {
            for (int v = tid; v < 8 * YLEN; v += 256) {
                const int tl = v >> 3, c8 = (v & 7) * 8;
                const int gt = t0 - 1 + tl;
                v8u val;
                if (gt >= 0 && gt < TLEN)
                    val = *reinterpret_cast<const v8u*>(
                        &y[((size_t)b * TLEN + gt) * CCH + kc + c8]);
                else {
#pragma unroll
                    for (int i = 0; i < 8; ++i) val[i] = 0;
                }
                *reinterpret_cast<v8u*>(&ysb[tl * KSTR + c8]) = val;
            }
        }
        for (int v = tid; v < 3 * 8 * OCT; v += 256) {
            const int tap = v >> 9;
            const int rem = v & 511;
            const int o = rem >> 3, c8 = (rem & 7) * 8;
            async_cp16(lds_off(&swa[bi][tap][o * WSS + c8]),
                       &wa[((size_t)tap * CCH * CCH) + (size_t)(obase + o) * CCH + kc + c8]);
        }
    };

    const int wv = tid >> 5, ln = tid & 31;
    const int l16 = ln & 15, half = ln >> 4;
    const int mt = wv & 3, nt0 = wv >> 2;

    const int wrow   = (mt * 16 + l16) * WSS;
    const int aoffLo = 16 * half;
    const int aoffHi = 16 * half + 8;
    const int boff   = 16 * half;
    int rowT[2];
#pragma unroll
    for (int p = 0; p < 2; ++p)
        rowT[p] = ((nt0 + 2 * p) * 16 + l16) * KSTR;

    v8f acc0, acc1;
#pragma unroll
    for (int i = 0; i < 8; ++i) { acc0[i] = 0.f; acc1[i] = 0.f; }

    issueChunk(0, 0);

#pragma unroll
    for (int ci = 0; ci < NCHUNK; ++ci) {
        const int bi = ci & 1;
        wait_async0();
        __syncthreads();
        if (ci + 1 < NCHUNK) issueChunk((ci + 1) * KC, bi ^ 1);

        const unsigned short* ysb = ys[bi];
#pragma unroll
        for (int kk = 0; kk < 2; ++kk) {
            const int kkb = kk * 32;
            const int bo = kkb + boff;
#pragma unroll
            for (int tap = 0; tap < 3; ++tap) {
                const unsigned short* wt = swa[bi][tap];
                const v16u a = cat8(lds8(wt + wrow + kkb + aoffLo), lds8(wt + wrow + kkb + aoffHi));
                const int rb0 = rowT[0] + tap * KSTR;
                const int rb1 = rowT[1] + tap * KSTR;
                const v16u b0 = cat8(lds8(ysb + rb0 + bo), lds8(ysb + rb0 + bo + 8));
                const v16u b1 = cat8(lds8(ysb + rb1 + bo), lds8(ysb + rb1 + bo + 8));
                acc0 = wmma_bf16(a, b0, acc0);
                acc1 = wmma_bf16(a, b1, acc1);
            }
        }
    }

    // epilogue: + bA + residual; fp32 store + bf16(lrelu) store for next layer
    const int ob8 = obase + mt * 16 + 8 * half;
    const float4 a0 = *reinterpret_cast<const float4*>(&bA[ob8]);
    const float4 a1 = *reinterpret_cast<const float4*>(&bA[ob8 + 4]);
    const float ba[8] = { a0.x, a0.y, a0.z, a0.w, a1.x, a1.y, a1.z, a1.w };
#pragma unroll
    for (int p = 0; p < 2; ++p) {
        const v8f acc = p ? acc1 : acc0;
        const int t = t0 + (nt0 + 2 * p) * 16 + l16;
        v8u pk;
#pragma unroll
        for (int r = 0; r < 8; ++r) {
            const size_t gi = ((size_t)b * CCH + ob8 + r) * TLEN + t;
            const float v = acc[r] + ba[r] + xin[gi];
            xout[gi] = v;
            pk[r] = f2bf(lrelu(v));
        }
        if (write_xbf)
            *reinterpret_cast<v8u*>(&xbf_out[((size_t)b * TLEN + t) * CCH + ob8]) = pk;
    }
}

// =====================================================================
extern "C" void kernel_launch(void* const* d_in, const int* in_sizes, int n_in,
                              void* d_out, int out_size, void* d_ws, size_t ws_size,
                              hipStream_t stream) {
    const float* x  = (const float*)d_in[0];
    const float* d  = (const float*)d_in[1];
    const float* WC = (const float*)d_in[2];
    const float* bC = (const float*)d_in[3];
    const float* WP = (const float*)d_in[4];
    const float* bP = (const float*)d_in[5];
    const float* WF = (const float*)d_in[6];
    const float* bF = (const float*)d_in[7];
    const float* WA = (const float*)d_in[8];
    const float* bA = (const float*)d_in[9];

    // workspace layout (bytes):
    //   y    bf16 [b][t][c] : 32 MB
    //   xbf  bf16 [b][t][c] : 32 MB
    //   x1   fp32 [b][c][t] : 64 MB
    //   bf16 weights        : ~2.4 MB
    char* ws = (char*)d_ws;
    const size_t SZ_BTC16 = (size_t)NB * TLEN * CCH * 2;
    unsigned short* yws  = (unsigned short*)(ws);
    unsigned short* xbf  = (unsigned short*)(ws + SZ_BTC16);
    float*          x1   = (float*)(ws + 2 * SZ_BTC16);
    unsigned short* wcb  = (unsigned short*)(ws + 2 * SZ_BTC16 + (size_t)NB * CCH * TLEN * 4);
    unsigned short* wpb  = wcb + (size_t)3 * 65536;
    unsigned short* wfb  = wpb + (size_t)3 * 65536;
    unsigned short* wab  = wfb + (size_t)3 * 65536;
    float* xo_final = (float*)d_out;

    prep_weights<<<dim3(3 * 65536 / 256), dim3(256), 0, stream>>>(WC, WP, WF, WA,
                                                                  wcb, wpb, wfb, wab);
    prep_x<<<dim3(TLEN / 64, CCH / 64, NB), dim3(256), 0, stream>>>(x, xbf);

    const int dil[3] = {1, 2, 4};
    dim3 grid(TLEN / TT, NB, CCH / OCT);   // (128, 8, 4)
    dim3 block(256);

    for (int i = 0; i < 3; ++i) {
        const size_t wOff = (size_t)i * 65536;
        const size_t bOff = (size_t)i * CCH;
        const float* xin = (i == 0) ? x : ((i == 1) ? x1 : xo_final);
        float* xout      = (i == 0) ? x1 : xo_final;   // i==2: in-place on d_out

        arb_stage1<<<grid, block, 0, stream>>>(
            xbf, d, wcb + wOff, wpb + wOff, wfb + wOff,
            bC + bOff, bP + bOff, bF + bOff, yws, dil[i]);

        arb_stage2<<<grid, block, 0, stream>>>(
            yws, wab + 3 * wOff, bA + bOff, xin, xout, xbf, (i < 2) ? 1 : 0);
    }
}
